// AttentiveRNNLanguageModel_70901320123000
// MI455X (gfx1250) — compile-verified, matrix operand order checked by reading, not run on hardware
//
#include <hip/hip_runtime.h>
#include <hip/hip_bf16.h>
#include <stdint.h>

#define BB 4
#define TT 1024
#define HH 512
#define PP 20
#define VV 32000

typedef __attribute__((ext_vector_type(16))) __bf16          bf16x16;
typedef __attribute__((ext_vector_type(8)))  float           f32x8;
typedef __attribute__((ext_vector_type(8)))  unsigned short  u16x8;

union FragU { bf16x16 bf; u16x8 h[2]; };

__device__ __forceinline__ unsigned short f2bf(float f) {
  unsigned int u = __builtin_bit_cast(unsigned int, f);
  u += 0x7FFFu + ((u >> 16) & 1u);          // round-to-nearest-even
  return (unsigned short)(u >> 16);
}
__device__ __forceinline__ float sigmf(float x) { return 1.0f / (1.0f + __expf(-x)); }

// Per-lane async global->LDS copy of 16 bytes (tracked by ASYNCcnt).
__device__ __forceinline__ void async_b128(const void* lds_dst, const void* gsrc) {
  const unsigned           la = (unsigned)(uintptr_t)lds_dst;  // LDS aperture low 32 bits
  const unsigned long long ga = (unsigned long long)(uintptr_t)gsrc;
  asm volatile("global_load_async_to_lds_b128 %0, %1, off"
               :: "v"(la), "v"(ga) : "memory");
}
__device__ __forceinline__ void wait_async0() {
  asm volatile("s_wait_asynccnt 0" ::: "memory");
}

// ---------------------------------------------------------------------------
// bf16 WMMA GEMM:  C[M,N](f32) = A[M,K](bf16 row-major) * Bt[N,K](bf16) + bias
// (Bt holds math-B transposed, i.e. C[m][n] = sum_k A[m][k]*Bt[n][k].)
// Block tile 128x128, BK=32, double-buffered async global->LDS staging.
// 8 waves: wave=(wr 0..3 -> 32 rows, wc 0..1 -> 64 cols); per k-chunk each
// wave loads 2 A-fragments + 4 B-fragments and issues 8 v_wmma (B reused 2x).
// Requires M%128==0, K%32==0.
// ---------------------------------------------------------------------------
__global__ __launch_bounds__(256)
void gemm_bf16_wmma(const unsigned short* __restrict__ A, long long batchA, int lda,
                    const unsigned short* __restrict__ Bt, long long batchB, int ldb,
                    float* __restrict__ C, long long batchC, int ldc,
                    int M, int N, int K, const float* __restrict__ bias)
{
  __shared__ unsigned short sA[2][128][32];   // A tile   [m][k]  (8 KB/buf)
  __shared__ unsigned short sB[2][128][32];   // Bt tile  [n][k]  (8 KB/buf)

  A  += (long long)blockIdx.z * batchA;
  Bt += (long long)blockIdx.z * batchB;
  C  += (long long)blockIdx.z * batchC;

  const int m0   = blockIdx.y * 128;
  const int n0   = blockIdx.x * 128;
  const int tid  = threadIdx.x;
  const int lane = tid & 31;
  const int wave = tid >> 5;
  const int wr   = wave & 3;      // 32-row strip 0..3
  const int wc   = wave >> 2;     // 64-col half 0..1
  const int half = lane >> 4;     // K-half select per ISA 16-bit A layout
  const int r16  = lane & 15;

  // staging coordinates (each thread: 2 A-chunks + 2 B-chunks of 16 B)
  const int ar = tid >> 2;            // 0..63  (rows ar and ar+64)
  const int ac = (tid & 3) * 8;       // 0,8,16,24
  const int br = tid >> 1;            // 0..127 (Bt row = n)
  const int bc = (tid & 1) * 16;      // 0 or 16

  const bool bInRange = (n0 + br) < N;

  f32x8 acc[8];                       // [sub(2) * 4 col-tiles]
  #pragma unroll
  for (int i = 0; i < 8; ++i)
    #pragma unroll
    for (int v = 0; v < 8; ++v) acc[i][v] = 0.0f;

  auto stage = [&](int buf, int k0) {
    async_b128(&sA[buf][ar][ac],      A + (long long)(m0 + ar) * lda + k0 + ac);
    async_b128(&sA[buf][ar + 64][ac], A + (long long)(m0 + ar + 64) * lda + k0 + ac);
    if (bInRange) {
      const unsigned short* src = Bt + (long long)(n0 + br) * ldb + k0 + bc;
      async_b128(&sB[buf][br][bc],     src);
      async_b128(&sB[buf][br][bc + 8], src + 8);
    } else {
      const u16x8 z = {0,0,0,0,0,0,0,0};
      *(u16x8*)&sB[buf][br][bc]     = z;
      *(u16x8*)&sB[buf][br][bc + 8] = z;
    }
  };

  stage(0, 0);
  wait_async0();
  __syncthreads();

  int buf = 0;
  for (int k0 = 0; k0 < K; k0 += 32) {
    if (k0 + 32 < K) stage(buf ^ 1, k0 + 32);   // overlap with compute below

    const int kb = half * 8;
    FragU a0, a1;
    const int arow = wr * 32 + r16;
    a0.h[0] = *(const u16x8*)&sA[buf][arow][kb];
    a0.h[1] = *(const u16x8*)&sA[buf][arow][kb + 16];
    a1.h[0] = *(const u16x8*)&sA[buf][arow + 16][kb];
    a1.h[1] = *(const u16x8*)&sA[buf][arow + 16][kb + 16];
    #pragma unroll
    for (int ct = 0; ct < 4; ++ct) {
      FragU b;
      const int bn = wc * 64 + ct * 16 + r16;
      b.h[0] = *(const u16x8*)&sB[buf][bn][kb];
      b.h[1] = *(const u16x8*)&sB[buf][bn][kb + 16];
      acc[ct] = __builtin_amdgcn_wmma_f32_16x16x32_bf16(
          false, a0.bf, false, b.bf, (short)0, acc[ct], false, false);
      acc[4 + ct] = __builtin_amdgcn_wmma_f32_16x16x32_bf16(
          false, a1.bf, false, b.bf, (short)0, acc[4 + ct], false, false);
    }

    wait_async0();          // our async stores to LDS (next buffer) complete
    __syncthreads();        // everyone's staging + fragment reads complete
    buf ^= 1;
  }

  // ---- epilogue: acc elem v -> row (sub*16 + half*8 + v), col r16 ----
  #pragma unroll
  for (int sub = 0; sub < 2; ++sub) {
    #pragma unroll
    for (int ct = 0; ct < 4; ++ct) {
      const int col = n0 + wc * 64 + ct * 16 + r16;
      if (col >= N) continue;
      const float bv = bias ? bias[col] : 0.0f;
      #pragma unroll
      for (int v = 0; v < 8; ++v) {
        const int row = m0 + wr * 32 + sub * 16 + half * 8 + v;
        if (row < M) C[(long long)row * ldc + col] = acc[sub * 4 + ct][v] + bv;
      }
    }
  }
}

// ---------------------------------------------------------------------------
// Encoder LSTM: one block per batch, 512 threads (thread = hidden unit),
// h kept in LDS, c in a register. xW precomputed by WMMA GEMM (incl. bih).
// ---------------------------------------------------------------------------
__global__ __launch_bounds__(512)
void lstm_enc_kernel(const float* __restrict__ xW, const float* __restrict__ Whh,
                     const float* __restrict__ bhh, float* __restrict__ enc)
{
  const int b = blockIdx.x;
  const int j = threadIdx.x;
  __shared__ float hsh[HH];
  float c = 0.0f;
  hsh[j] = 0.0f;
  __syncthreads();
  const float* xWb = xW + (long long)b * TT * (4 * HH);
  const float* w0 = Whh + (long long)(0 * HH + j) * HH;
  const float* w1 = Whh + (long long)(1 * HH + j) * HH;
  const float* w2 = Whh + (long long)(2 * HH + j) * HH;
  const float* w3 = Whh + (long long)(3 * HH + j) * HH;
  const float c0 = bhh[0 * HH + j], c1 = bhh[1 * HH + j];
  const float c2 = bhh[2 * HH + j], c3 = bhh[3 * HH + j];
  for (int t = 0; t < TT; ++t) {
    const float* xt = xWb + (long long)t * (4 * HH);
    float a0 = xt[0 * HH + j] + c0;
    float a1 = xt[1 * HH + j] + c1;
    float a2 = xt[2 * HH + j] + c2;
    float a3 = xt[3 * HH + j] + c3;
    #pragma unroll 4
    for (int k = 0; k < HH; ++k) {
      const float hk = hsh[k];
      a0 = fmaf(w0[k], hk, a0);
      a1 = fmaf(w1[k], hk, a1);
      a2 = fmaf(w2[k], hk, a2);
      a3 = fmaf(w3[k], hk, a3);
    }
    const float ig = sigmf(a0), fg = sigmf(a1), gg = tanhf(a2), og = sigmf(a3);
    c = fg * c + ig * gg;
    const float h = og * tanhf(c);
    __syncthreads();
    hsh[j] = h;
    enc[((long long)b * TT + t) * HH + j] = h;
    __syncthreads();
  }
}

// ---------------------------------------------------------------------------
// Positional LSTM (P=20): one block, thread = (batch, unit). pxW is [B,T,80].
// ---------------------------------------------------------------------------
__global__ __launch_bounds__(128)
void lstm_pos_kernel(const float* __restrict__ pxW, const float* __restrict__ Wp_hh,
                     const float* __restrict__ bp_ih, const float* __restrict__ bp_hh,
                     float* __restrict__ pw)
{
  const int tid = threadIdx.x;
  const bool act = tid < BB * PP;
  const int b = tid / PP, j = tid % PP;
  __shared__ float hp[BB * PP];
  float c = 0.0f;
  if (act) hp[tid] = 0.0f;
  __syncthreads();
  for (int t = 0; t < TT; ++t) {
    float g[4] = {0.f, 0.f, 0.f, 0.f};
    if (act) {
      #pragma unroll
      for (int q = 0; q < 4; ++q) {
        const int r = q * PP + j;
        float a = pxW[((long long)b * TT + t) * (4 * PP) + r] + bp_ih[r] + bp_hh[r];
        #pragma unroll
        for (int k = 0; k < PP; ++k) a = fmaf(Wp_hh[r * PP + k], hp[b * PP + k], a);
        g[q] = a;
      }
    }
    __syncthreads();
    if (act) {
      const float ig = sigmf(g[0]), fg = sigmf(g[1]), gg = tanhf(g[2]), og = sigmf(g[3]);
      c = fg * c + ig * gg;
      const float h = og * tanhf(c);
      hp[tid] = h;
      pw[((long long)b * TT + t) * PP + j] = h;
    }
    __syncthreads();
  }
}

// ---------------------------------------------------------------------------
// Heads: mw (relu), sigma (sigmoid), base term.
// ---------------------------------------------------------------------------
__global__ void head_kernel(const float* __restrict__ pw, const float* __restrict__ Wmu,
                            const float* __restrict__ bmu, const float* __restrict__ Wsig,
                            const float* __restrict__ bsig, const int* __restrict__ padL,
                            float* __restrict__ a0, float* __restrict__ base,
                            float* __restrict__ sigma)
{
  const int idx = blockIdx.x * blockDim.x + threadIdx.x;
  if (idx >= BB * TT) return;
  const int b = idx / TT, t = idx % TT;
  const float* p = pw + (long long)idx * PP;
  float d0 = bmu[0], d1 = bmu[1], d2 = bmu[2], dsg = bsig[0];
  #pragma unroll
  for (int k = 0; k < PP; ++k) {
    const float pv = p[k];
    d0 = fmaf(Wmu[0 * PP + k], pv, d0);
    d1 = fmaf(Wmu[1 * PP + k], pv, d1);
    d2 = fmaf(Wmu[2 * PP + k], pv, d2);
    dsg = fmaf(Wsig[k], pv, dsg);
  }
  d0 = fmaxf(d0, 0.0f); d1 = fmaxf(d1, 0.0f); d2 = fmaxf(d2, 0.0f);
  const float invL = 1.0f / (float)padL[b];
  a0[idx] = d0;
  sigma[idx] = sigmf(dsg);
  base[idx] = d1 * invL + d2 * ((float)(t + 1)) * invL;
}

__global__ void mu_scan_kernel(const float* __restrict__ a0, const float* __restrict__ base,
                               float* __restrict__ mu)
{
  const int b = threadIdx.x;
  if (b >= BB) return;
  float m = 0.0f;
  for (int t = 0; t < TT; ++t) {
    const int i = b * TT + t;
    m = a0[i] * m + base[i];
    mu[i] = m;
  }
}

// ---------------------------------------------------------------------------
// Attention row: gaussian weights, causal mask, L1 normalize, emit bf16.
// ---------------------------------------------------------------------------
__global__ __launch_bounds__(256)
void attn_kernel(const float* __restrict__ mu, const float* __restrict__ sigma,
                 unsigned short* __restrict__ w)
{
  const int idx = blockIdx.x;            // b*T + j
  const int j = idx % TT;
  const float m = mu[idx], s = sigma[idx];
  const float inv2s = 1.0f / (2.0f * s * s + 0.001f);
  const float jinv = 1.0f / (float)(j + 1);
  __shared__ float row[TT];
  __shared__ float ssum;
  if (threadIdx.x == 0) ssum = 0.0f;
  __syncthreads();
  float part = 0.0f;
  for (int t = threadIdx.x; t < TT; t += 256) {
    float e = 0.0f;
    if (t <= j) { const float d = (float)t * jinv - m; e = __expf(-d * d * inv2s); }
    row[t] = e;
    part += e;
  }
  atomicAdd(&ssum, part);                // ds_add_f32
  __syncthreads();
  const float inv = 1.0f / fmaxf(ssum, 1e-12f);
  unsigned short* wr = w + (long long)idx * TT;
  for (int t = threadIdx.x; t < TT; t += 256) wr[t] = f2bf(row[t] * inv);
}

// ---------------------------------------------------------------------------
// Small utility kernels
// ---------------------------------------------------------------------------
__global__ void gather_kernel(const int* __restrict__ ids, const float* __restrict__ emb,
                              unsigned short* __restrict__ xbf)
{
  const long long e = (long long)blockIdx.x * blockDim.x + threadIdx.x;
  if (e >= (long long)BB * TT * HH) return;
  const int row = (int)(e >> 9);
  const int col = (int)(e & 511);
  xbf[e] = f2bf(emb[(long long)ids[row] * HH + col]);
}

// plain f32 -> bf16 convert (layout preserved)
__global__ void conv_kernel(const float* __restrict__ src, unsigned short* __restrict__ dst,
                            long long n)
{
  const long long e = (long long)blockIdx.x * blockDim.x + threadIdx.x;
  if (e >= n) return;
  dst[e] = f2bf(src[e]);
}

// enc [B,T,H] f32 -> encT [B,H,T] bf16 (for ctx GEMM B-operand, [N][K] layout)
__global__ void encT_kernel(const float* __restrict__ enc, unsigned short* __restrict__ encT)
{
  const long long e = (long long)blockIdx.x * blockDim.x + threadIdx.x;
  if (e >= (long long)BB * HH * TT) return;
  const int b = (int)(e / (HH * TT));
  const int r = (int)(e % (HH * TT));
  const int h = r / TT, t = r % TT;
  encT[e] = f2bf(enc[((long long)b * TT + t) * HH + h]);
}

// strided f32 -> bf16 with dest column offset (builds concat [ctx, enc])
__global__ void convrows_kernel(const float* __restrict__ src, int ldsrc,
                                unsigned short* __restrict__ dst, int lddst, int colOff,
                                long long rows, int cols)
{
  const long long e = (long long)blockIdx.x * blockDim.x + threadIdx.x;
  if (e >= rows * (long long)cols) return;
  const long long m = e / cols;
  const int c = (int)(e % cols);
  dst[m * lddst + colOff + c] = f2bf(src[m * ldsrc + c]);
}

__global__ void tanhconv_kernel(const float* __restrict__ src, unsigned short* __restrict__ dst,
                                long long n)
{
  const long long e = (long long)blockIdx.x * blockDim.x + threadIdx.x;
  if (e >= n) return;
  dst[e] = f2bf(tanhf(src[e]));
}

// ---------------------------------------------------------------------------
extern "C" void kernel_launch(void* const* d_in, const int* in_sizes, int n_in,
                              void* d_out, int out_size, void* d_ws, size_t ws_size,
                              hipStream_t stream)
{
  (void)in_sizes; (void)n_in; (void)out_size; (void)ws_size;
  const int*   input_ids = (const int*)d_in[0];
  const int*   pad_len   = (const int*)d_in[1];
  const float* emb       = (const float*)d_in[2];
  const float* dec_bias  = (const float*)d_in[3];
  const float* Wih       = (const float*)d_in[4];
  const float* Whh       = (const float*)d_in[5];
  const float* bih       = (const float*)d_in[6];
  const float* bhh       = (const float*)d_in[7];
  const float* Wp_ih     = (const float*)d_in[8];
  const float* Wp_hh     = (const float*)d_in[9];
  const float* bp_ih     = (const float*)d_in[10];
  const float* bp_hh     = (const float*)d_in[11];
  const float* Wmu       = (const float*)d_in[12];
  const float* bmu       = (const float*)d_in[13];
  const float* Wsig      = (const float*)d_in[14];
  const float* bsig      = (const float*)d_in[15];
  const float* Wc        = (const float*)d_in[16];
  const float* bc        = (const float*)d_in[17];
  float* out = (float*)d_out;

  char* ws = (char*)d_ws;
  size_t off = 0;
  auto alloc = [&](size_t bytes) -> char* {
    char* p = ws + off;
    off = (off + bytes + 255) & ~(size_t)255;
    return p;
  };
  unsigned short* embbf  = (unsigned short*)alloc((size_t)VV * HH * 2);       // [V,H] = Bt
  unsigned short* Wihbf  = (unsigned short*)alloc((size_t)4 * HH * HH * 2);   // [4H,H] = Bt
  unsigned short* Wpibf  = (unsigned short*)alloc((size_t)4 * PP * HH * 2);   // [80,512] = Bt
  unsigned short* Wcbf   = (unsigned short*)alloc((size_t)HH * 2 * HH * 2);   // [512,1024] = Bt
  unsigned short* xbf    = (unsigned short*)alloc((size_t)BB * TT * HH * 2);
  float*          xW     = (float*)alloc((size_t)BB * TT * 4 * HH * 4);
  float*          enc    = (float*)alloc((size_t)BB * TT * HH * 4);
  unsigned short* encbf  = (unsigned short*)alloc((size_t)BB * TT * HH * 2);  // [B*T,H] A-op
  unsigned short* encTbf = (unsigned short*)alloc((size_t)BB * HH * TT * 2);  // [B,H,T] Bt-op
  float*          pxW    = (float*)alloc((size_t)BB * TT * 4 * PP * 4);
  float*          pwbuf  = (float*)alloc((size_t)BB * TT * PP * 4);
  float*          a0b    = (float*)alloc((size_t)BB * TT * 4);
  float*          baseb  = (float*)alloc((size_t)BB * TT * 4);
  float*          sigb   = (float*)alloc((size_t)BB * TT * 4);
  float*          mub    = (float*)alloc((size_t)BB * TT * 4);
  unsigned short* wbf    = (unsigned short*)alloc((size_t)BB * TT * TT * 2);
  float*          ctx    = (float*)alloc((size_t)BB * TT * HH * 4);
  unsigned short* catbf  = (unsigned short*)alloc((size_t)BB * TT * 2 * HH * 2);
  float*          comb   = (float*)alloc((size_t)BB * TT * HH * 4);
  unsigned short* combbf = (unsigned short*)alloc((size_t)BB * TT * HH * 2);

  const int MT = BB * TT;   // 4096

  // 1) weight prep: plain bf16 conversion (weights are already [N][K])
  conv_kernel<<<dim3((unsigned)(((size_t)VV * HH + 255) / 256)), 256, 0, stream>>>(emb, embbf, (long long)VV * HH);
  conv_kernel<<<dim3((4 * HH * HH + 255) / 256), 256, 0, stream>>>(Wih, Wihbf, (long long)4 * HH * HH);
  conv_kernel<<<dim3((4 * PP * HH + 255) / 256), 256, 0, stream>>>(Wp_ih, Wpibf, (long long)4 * PP * HH);
  conv_kernel<<<dim3((HH * 2 * HH + 255) / 256), 256, 0, stream>>>(Wc, Wcbf, (long long)HH * 2 * HH);

  // 2) embedding gather -> bf16
  gather_kernel<<<dim3((MT * HH + 255) / 256), 256, 0, stream>>>(input_ids, emb, xbf);

  // 3) xW = x @ Wih^T + bih   [4096 x 2048]
  gemm_bf16_wmma<<<dim3(4 * HH / 128, MT / 128, 1), 256, 0, stream>>>(
      xbf, 0, HH, Wihbf, 0, HH, xW, 0, 4 * HH, MT, 4 * HH, HH, bih);

  // 4) encoder LSTM (sequential)
  lstm_enc_kernel<<<dim3(BB), 512, 0, stream>>>(xW, Whh, bhh, enc);

  // 5) enc -> bf16 (A layout) and transposed bf16 (Bt layout for ctx GEMM)
  conv_kernel<<<dim3((MT * HH + 255) / 256), 256, 0, stream>>>(enc, encbf, (long long)MT * HH);
  encT_kernel<<<dim3((MT * HH + 255) / 256), 256, 0, stream>>>(enc, encTbf);

  // 6) pxW = enc @ Wp_ih^T   [4096 x 80] (biases added inside pos-LSTM)
  gemm_bf16_wmma<<<dim3(1, MT / 128, 1), 256, 0, stream>>>(
      encbf, 0, HH, Wpibf, 0, HH, pxW, 0, 4 * PP, MT, 4 * PP, HH, nullptr);

  // 7) positional LSTM (sequential)
  lstm_pos_kernel<<<dim3(1), 128, 0, stream>>>(pxW, Wp_hh, bp_ih, bp_hh, pwbuf);

  // 8) heads + mu scan
  head_kernel<<<dim3((MT + 255) / 256), 256, 0, stream>>>(pwbuf, Wmu, bmu, Wsig, bsig,
                                                          pad_len, a0b, baseb, sigb);
  mu_scan_kernel<<<dim3(1), 32, 0, stream>>>(a0b, baseb, mub);

  // 9) attention weights -> bf16 [B,T,T]
  attn_kernel<<<dim3(MT), 256, 0, stream>>>(mub, sigb, wbf);

  // 10) ctx = w @ enc (batched over B); Bt = encT [H,T]
  gemm_bf16_wmma<<<dim3(HH / 128, TT / 128, BB), 256, 0, stream>>>(
      wbf, (long long)TT * TT, TT, encTbf, (long long)HH * TT, TT,
      ctx, (long long)TT * HH, HH, TT, HH, TT, nullptr);

  // 11) concat [ctx, enc] -> bf16 [4096 x 1024]
  convrows_kernel<<<dim3((MT * HH + 255) / 256), 256, 0, stream>>>(ctx, HH, catbf, 2 * HH, 0, MT, HH);
  convrows_kernel<<<dim3((MT * HH + 255) / 256), 256, 0, stream>>>(enc, HH, catbf, 2 * HH, HH, MT, HH);

  // 12) combined = cat @ Wc^T + bc, then tanh -> bf16
  gemm_bf16_wmma<<<dim3(HH / 128, MT / 128, 1), 256, 0, stream>>>(
      catbf, 0, 2 * HH, Wcbf, 0, 2 * HH, comb, 0, HH, MT, HH, 2 * HH, bc);
  tanhconv_kernel<<<dim3((MT * HH + 255) / 256), 256, 0, stream>>>(comb, combbf, (long long)MT * HH);

  // 13) logits = combined @ emb^T + dec_bias  [4096 x 32000] -> d_out
  gemm_bf16_wmma<<<dim3((unsigned)(VV / 128), MT / 128, 1), 256, 0, stream>>>(
      combbf, 0, HH, embbf, 0, HH, out, 0, VV, MT, VV, HH, dec_bias);
}
